// PalmBridge_7000796692912
// MI455X (gfx1250) — matrix operand order repaired and checked by reading, not compiled, not run on hardware
//
#include <hip/hip_runtime.h>

#define DDIM 512
#define KCB  512
#define NROWS 131072
#define ROWS_PER_BLOCK 128
#define THREADS 256
#define ZS 520        // padded LDS stride (bf16 elems): 1040B = 16*65, shifts banks by 4/row
#define BSTRIDE 520

typedef __bf16 bf16_t;
typedef __bf16 v16bf __attribute__((ext_vector_type(16)));
typedef __bf16 v8bf  __attribute__((ext_vector_type(8)));
typedef float  v8f   __attribute__((ext_vector_type(8)));

union V16 { v16bf v; v8bf h[2]; };

// ---------------- prep: P (f32) -> Pb (bf16), pnorm = ||P_k||^2 (f32) ----------------
__global__ __launch_bounds__(256) void vq_prep(const float* __restrict__ P,
                                               bf16_t* __restrict__ Pb,
                                               float* __restrict__ pnorm) {
    __shared__ float red[256];
    const int k = blockIdx.x;
    const int t = threadIdx.x;
    const float2 v = ((const float2*)(P + (size_t)k * DDIM))[t];   // 256 thr * 2 = 512
    union { bf16_t h[2]; unsigned u; } pk;
    pk.h[0] = (bf16_t)v.x;
    pk.h[1] = (bf16_t)v.y;
    ((unsigned*)(Pb + (size_t)k * DDIM))[t] = pk.u;
    red[t] = v.x * v.x + v.y * v.y;
    __syncthreads();
    for (int s = 128; s > 0; s >>= 1) {
        if (t < s) red[t] += red[t + s];
        __syncthreads();
    }
    if (t == 0) pnorm[k] = red[0];
}

// ---------------- main: bf16 WMMA distances + argmin + f32 gather/blend ----------------
__global__ __launch_bounds__(THREADS) void vq_main(const float* __restrict__ z,
                                                   const float* __restrict__ P,
                                                   const bf16_t* __restrict__ Pb,
                                                   const float* __restrict__ pnorm,
                                                   float* __restrict__ out) {
    extern __shared__ char smem[];
    bf16_t* zl = (bf16_t*)smem;                                        // 128 x ZS bf16
    bf16_t* bl = (bf16_t*)(smem + ROWS_PER_BLOCK * ZS * 2);            // 64 x BSTRIDE bf16
    int*    si = (int*)(smem + ROWS_PER_BLOCK * ZS * 2 + 64 * BSTRIDE * 2);

    const int tid  = threadIdx.x;
    const int lane = tid & 31;
    const int wave = tid >> 5;
    const int half = lane >> 4;
    const int l16  = lane & 15;
    const int rowBase = blockIdx.x * ROWS_PER_BLOCK;

    // ---- stage 128 z rows into LDS as bf16 (8 floats / iter / thread) ----
    for (int i = tid; i < ROWS_PER_BLOCK * (DDIM / 8); i += THREADS) {
        const int r  = i >> 6;             // 64 8-elem chunks per row
        const int c8 = (i & 63) * 8;
        const float4* src = (const float4*)(z + (size_t)(rowBase + r) * DDIM + c8);
        const float4 f0 = src[0];
        const float4 f1 = src[1];
        union { bf16_t h[8]; uint4 u; } pk;
        pk.h[0] = (bf16_t)f0.x; pk.h[1] = (bf16_t)f0.y;
        pk.h[2] = (bf16_t)f0.z; pk.h[3] = (bf16_t)f0.w;
        pk.h[4] = (bf16_t)f1.x; pk.h[5] = (bf16_t)f1.y;
        pk.h[6] = (bf16_t)f1.z; pk.h[7] = (bf16_t)f1.w;
        *(uint4*)(zl + r * ZS + c8) = pk.u;
    }
    __syncthreads();

    float bestv[8];
    int   besti[8];
#pragma unroll
    for (int r = 0; r < 8; ++r) { bestv[r] = 3.4e38f; besti[r] = 0; }

    // A-fragment base for this lane: row = wave*16 + l16, K-chunk offset by lane half
    const bf16_t* arow = zl + (wave * 16 + l16) * ZS + half * 8;

    for (int cg = 0; cg < 8; ++cg) {
        // ---- stage 64 codebook columns (bf16 rows of Pb) into LDS ----
        __syncthreads();   // previous group's B reads complete
        for (int i = tid; i < 64 * (DDIM / 8); i += THREADS) {
            const int c  = i >> 6;
            const int s8 = (i & 63) * 8;
            *(uint4*)(bl + c * BSTRIDE + s8) =
                *(const uint4*)(Pb + (size_t)(cg * 64 + c) * DDIM + s8);
        }
        __syncthreads();

        v8f acc[4];
#pragma unroll
        for (int t = 0; t < 4; ++t) acc[t] = (v8f){0.f,0.f,0.f,0.f,0.f,0.f,0.f,0.f};

#pragma unroll 4
        for (int kb = 0; kb < 16; ++kb) {
            // A 16x32 bf16: lanes 0-15 hold K {0..7,16..23}, lanes 16-31 hold K {8..15,24..31}
            V16 A;
            A.h[0] = *(const v8bf*)(arow + kb * 32);
            A.h[1] = *(const v8bf*)(arow + kb * 32 + 16);
#pragma unroll
            for (int t = 0; t < 4; ++t) {
                // B 32x16 bf16: col = lane%16, K {0..7,16..23} (lanes 0-15) / {8..15,24..31} (lanes 16-31)
                const bf16_t* bp = bl + (t * 16 + l16) * BSTRIDE + kb * 32 + half * 8;
                V16 B;
                B.h[0] = *(const v8bf*)(bp);
                B.h[1] = *(const v8bf*)(bp + 16);
                acc[t] = __builtin_amdgcn_wmma_f32_16x16x32_bf16(
                             false, A.v, false, B.v, (short)0, acc[t], false, false);
            }
        }

        // dist = ||p||^2 - 2 z.p  (||z||^2 is per-row constant -> irrelevant to argmin)
#pragma unroll
        for (int t = 0; t < 4; ++t) {
            const int col = (cg * 4 + t) * 16 + l16;
            const float pn = pnorm[col];
#pragma unroll
            for (int r = 0; r < 8; ++r) {
                const float d = pn - 2.0f * acc[t][r];
                if (d < bestv[r]) { bestv[r] = d; besti[r] = col; }   // strict < => first min
            }
        }
    }

    // ---- cross-lane argmin within each 16-lane half (rows r / r+8) ----
#pragma unroll
    for (int r = 0; r < 8; ++r) {
        float v = bestv[r];
        int  bi = besti[r];
#pragma unroll
        for (int m = 1; m < 16; m <<= 1) {
            const float ov = __shfl_xor(v, m, 32);
            const int   oi = __shfl_xor(bi, m, 32);
            if (ov < v || (ov == v && oi < bi)) { v = ov; bi = oi; }
        }
        besti[r] = bi;
    }
    if (l16 == 0) {
#pragma unroll
        for (int r = 0; r < 8; ++r) si[wave * 16 + half * 8 + r] = besti[r];
    }
    __syncthreads();

    float* zhat = out;
    float* ztil = out + (size_t)NROWS * DDIM;
    float* idxo = out + (size_t)2 * NROWS * DDIM;
    if (tid < ROWS_PER_BLOCK) idxo[rowBase + tid] = (float)si[tid];

    // ---- f32 gather + blend (z re-read hits L2; outputs streamed) ----
#pragma unroll 1
    for (int rr = 0; rr < 16; ++rr) {
        const int lrow = wave * 16 + rr;
        const int grow = rowBase + lrow;
        const int id   = si[lrow];
        const float4* zp = (const float4*)(z + (size_t)grow * DDIM);
        const float4* pp = (const float4*)(P + (size_t)id * DDIM);
        float4* zhp = (float4*)(zhat + (size_t)grow * DDIM);
        float4* ztp = (float4*)(ztil + (size_t)grow * DDIM);
        for (int j = lane; j < DDIM / 4; j += 32) {
            const float4 zv = zp[j];
            const float4 pv = pp[j];
            float4 zh;
            zh.x = 0.7f * zv.x + 0.3f * pv.x;
            zh.y = 0.7f * zv.y + 0.3f * pv.y;
            zh.z = 0.7f * zv.z + 0.3f * pv.z;
            zh.w = 0.7f * zv.w + 0.3f * pv.w;
            ztp[j] = pv;
            zhp[j] = zh;
        }
    }
}

extern "C" void kernel_launch(void* const* d_in, const int* in_sizes, int n_in,
                              void* d_out, int out_size, void* d_ws, size_t ws_size,
                              hipStream_t stream) {
    const float* z = (const float*)d_in[0];   // [N, D] f32
    const float* P = (const float*)d_in[1];   // [K, D] f32

    bf16_t* Pb    = (bf16_t*)d_ws;                                  // K*D bf16 = 512 KB
    float*  pnorm = (float*)((char*)d_ws + (size_t)KCB * DDIM * 2); // K f32

    vq_prep<<<KCB, 256, 0, stream>>>(P, Pb, pnorm);

    const size_t shmem = (size_t)ROWS_PER_BLOCK * ZS * 2   // z tile  (133120 B)
                       + (size_t)64 * BSTRIDE * 2          // B tile  ( 66560 B)
                       + (size_t)ROWS_PER_BLOCK * sizeof(int); // idx  (   512 B)
    vq_main<<<NROWS / ROWS_PER_BLOCK, THREADS, shmem, stream>>>(
        z, P, Pb, pnorm, (float*)d_out);
}
